// AttentionModule_7438883356929
// MI455X (gfx1250) — matrix operand order repaired and checked by reading, not compiled, not run on hardware
//
#include <hip/hip_runtime.h>
#include <hip/hip_bf16.h>

typedef __attribute__((ext_vector_type(16))) __bf16 v16bf;
typedef __attribute__((ext_vector_type(8)))  float  v8f;
typedef __attribute__((ext_vector_type(4)))  unsigned int v4u;
typedef __attribute__((ext_vector_type(8)))  int v8i;
typedef __attribute__((ext_vector_type(4)))  int v4i;

struct __align__(16) bf8 { __bf16 h[8]; };

#define B_SZ   4
#define C_IN   128
#define N_PTS  4096
#define K_GRP  32
#define NK     (N_PTS * K_GRP)     // 131072 spatial per batch
#define NGRP   32
#define GN_EPS 1e-5f

// Toolchain discriminator (probe-verified): the gfx1250 TDM header ships only
// with the 6-arg tensor_load_to_lds toolchain.
#if __has_include(<hip/amd_detail/amd_gfx1250_TDM.h>)
#define TDM_6ARG 1
#else
#define TDM_6ARG 0
#endif
#define USE_TDM 1

// ---------------------------------------------------------------------------
// Swizzled fragment layout: one WMMA B-fragment (32ci x 16p tile) is 32
// contiguous lane-slots of 16 bf16:
//   idx(c,p) = ((c>>5)*(S>>4) + (p>>4))*512 + ((p&15) + (((c>>4)&1)<<4))*16 + (c&15)
// Consumer lane L reads 16 contiguous bf16 (two b128); producer D-tiles write
// 8 contiguous bf16 (one b128).
// ---------------------------------------------------------------------------
static __device__ inline void load_frag(const __bf16* p, v16bf& f) {
    bf8 a0 = *(const bf8*)p;
    bf8 a1 = *(const bf8*)(p + 8);
#pragma unroll
    for (int i = 0; i < 8; ++i) { f[i] = a0.h[i]; f[8 + i] = a1.h[i]; }
}

static __device__ inline v8f wmma_bf16(const v16bf& a, const v16bf& b, const v8f& c) {
    return __builtin_amdgcn_wmma_f32_16x16x32_bf16(false, a, false, b, (short)0, c, false, false);
}

// A-fragment source-channel mapping (shared by the two swizzle kernels):
// frag idx = ((o0/16)*4 + cb/32)*512 + lane*16 + e
static __device__ inline void afrag_src(int idx, int& row, int& ci) {
    int e = idx & 15, lane = (idx >> 4) & 31, ct = (idx >> 9) & 3, ot = idx >> 11;
    int m = lane & 15, kk = (lane < 16) ? 0 : 16;
    int K = kk + e;
    row = ot * 16 + m;
    ci  = ct * 32 + K;
}

// Pre-swizzle fp32 weights into bf16 WMMA A-fragment order (no folding).
__global__ void wswz_prep(const float* __restrict__ W, __bf16* __restrict__ dst, int total) {
    int idx = blockIdx.x * 256 + threadIdx.x;
    if (idx >= total) return;
    int row, ci; afrag_src(idx, row, ci);
    dst[idx] = (__bf16)W[(size_t)row * C_IN + ci];
}

// Per-batch GN-folded swizzled weights: W' = W * (rstd_g * gamma_c).
__global__ void fold_wswz(const float* __restrict__ W, const float* __restrict__ rstd,
                          const float* __restrict__ gnw, __bf16* __restrict__ dst) {
    int b = blockIdx.y;
    int idx = blockIdx.x * 256 + threadIdx.x;        // 0..16383
    int row, ci; afrag_src(idx, row, ci);
    float sc = rstd[b * NGRP + (ci >> 2)] * gnw[ci];
    dst[(size_t)b * (C_IN * C_IN) + idx] = (__bf16)(W[(size_t)row * C_IN + ci] * sc);
}

// Per-batch folded bias: b' = b + W @ (beta - mu*rstd*gamma).
__global__ void fold_bias(const float* __restrict__ W, const float* __restrict__ bias,
                          const float* __restrict__ mean, const float* __restrict__ rstd,
                          const float* __restrict__ gnw, const float* __restrict__ gnb,
                          float* __restrict__ bf) {
    int b = blockIdx.y, o = threadIdx.x;             // 128 threads
    float acc = bias[o];
    for (int ci = 0; ci < C_IN; ++ci) {
        float sc = rstd[b * NGRP + (ci >> 2)] * gnw[ci];
        float t  = gnb[ci] - mean[b * NGRP + (ci >> 2)] * sc;
        acc += W[(size_t)o * C_IN + ci] * t;
    }
    bf[b * C_IN + o] = acc;
}

// ---------------------------------------------------------------------------
// conv1x1 (fp32 in, LDS-staged, non-temporal) + bias (+ReLU) + GN partials.
// MODE: 0 = swizzled bf16 (WMMA-B order), 1 = (B,N,64) c-contig, 2 = raw (B,C,S).
// ---------------------------------------------------------------------------
template<int COUT, bool RELU, int MODE>
__global__ void __launch_bounds__(256) conv1x1_gn(
    const float* __restrict__ X, const __bf16* __restrict__ Wswz,
    const float* __restrict__ bias, __bf16* __restrict__ Yout,
    float* __restrict__ gnAcc, int spatial, int chanOff, float sumw)
{
    constexpr int CO_TILES = COUT / 16;
    constexpr int SP_TILES = 8 / CO_TILES;
    constexpr int SPB      = SP_TILES * 16;
    constexpr int RPP      = 256 / SPB;
    constexpr int PASSES   = C_IN / RPP;
    __shared__ __bf16 sX[4 * SP_TILES * 512];
    __shared__ float s_sum[NGRP], s_sq[NGRP];
    int tid = threadIdx.x, lane = tid & 31, wv = tid >> 5;
    if (tid < NGRP) { s_sum[tid] = 0.f; s_sq[tid] = 0.f; }
    int b = blockIdx.y;
    size_t pblk = (size_t)blockIdx.x * SPB;
    const float* xb = X + (size_t)b * C_IN * (size_t)spatial;

    int colg = tid & (SPB - 1);
    int rbase = tid / SPB;
    int tp = colg >> 4, colc = colg & 15;
#pragma unroll
    for (int ps = 0; ps < PASSES; ++ps) {
        int ci = ps * RPP + rbase;
        float v = __builtin_nontemporal_load(&xb[(size_t)ci * spatial + pblk + colg]);
        int slot = colc + (((ci >> 4) & 1) << 4);
        sX[((ci >> 5) * SP_TILES + tp) * 512 + slot * 16 + (ci & 15)] = (__bf16)v;
    }
    __syncthreads();

    int o0 = (wv % CO_TILES) * 16;
    int spt = wv / CO_TILES;
    const __bf16* wbase = Wswz + (size_t)(o0 >> 4) * 4 * 512;
    v8f acc = {};
#pragma unroll
    for (int ct = 0; ct < 4; ++ct) {
        v16bf a, bm;
        load_frag(wbase + ct * 512 + lane * 16, a);
        load_frag(&sX[(ct * SP_TILES + spt) * 512 + lane * 16], bm);
        acc = wmma_bf16(a, bm, acc);
    }

    int m_off = (lane < 16) ? 0 : 8, col = lane & 15;
    size_t p = pblk + spt * 16 + col;
    bf8 dv;
#pragma unroll
    for (int v = 0; v < 8; ++v) {
        int c = o0 + v + m_off;
        float val = acc[v] + bias[c];
        if (RELU) val = fmaxf(val, 0.f);
        dv.h[v] = (__bf16)val;
        int g = (chanOff + c) >> 2;
        atomicAdd(&s_sum[g], val * sumw);
        atomicAdd(&s_sq[g],  val * val * sumw);
    }
    __bf16* yb = Yout + (size_t)b * COUT * (size_t)spatial;
    if (MODE == 0) {
        size_t idx = ((size_t)(o0 >> 5) * ((size_t)spatial >> 4) + (p >> 4)) * 512
                   + (size_t)(col + (((o0 >> 4) & 1) << 4)) * 16 + m_off;
        *(bf8*)&yb[idx] = dv;
    } else if (MODE == 1) {
        *(bf8*)&Yout[((size_t)b * spatial + p) * 64 + o0 + m_off] = dv;
    } else {
#pragma unroll
        for (int v = 0; v < 8; ++v)
            yb[(size_t)(o0 + v + m_off) * spatial + p] = dv.h[v];
    }
    __syncthreads();
    if (tid < NGRP) {
        atomicAdd(&gnAcc[(b * NGRP + tid) * 2 + 0], s_sum[tid]);
        atomicAdd(&gnAcc[(b * NGRP + tid) * 2 + 1], s_sq[tid]);
    }
}

__global__ void zero_acc(float* a1, float* a2, float* a3) {
    int t = threadIdx.x;
    a1[t] = 0.f; a2[t] = 0.f; a3[t] = 0.f;
}

__global__ void gn_finalize(const float* __restrict__ acc, float* __restrict__ mean,
                            float* __restrict__ rstd, float invCnt) {
    int i = threadIdx.x;               // 128 = B*NGRP
    float s = acc[2 * i], q = acc[2 * i + 1];
    float m = s * invCnt;
    float var = q * invCnt - m * m;
    mean[i] = m;
    rstd[i] = rsqrtf(var + GN_EPS);
}

// ---------------------------------------------------------------------------
// z2 = relu(W' @ concat(Yfeat bcast, Yg) + b') with GN1 folded into W'/b'.
// Raw bf16 fragments -> WMMA directly. GN2 partials in epilogue.
// ---------------------------------------------------------------------------
__global__ void __launch_bounds__(256) wc1_kernel(
    const __bf16* __restrict__ YfeatT, const __bf16* __restrict__ YgSwz,
    const __bf16* __restrict__ Wfold, const float* __restrict__ bfold,
    __bf16* __restrict__ Z2swz, float* __restrict__ gnAcc2)
{
    __shared__ __bf16 sYf[64];
    __shared__ __bf16 sYg[1024];
    __shared__ float s_sum[NGRP], s_sq[NGRP];
    int tid = threadIdx.x, lane = tid & 31, wv = tid >> 5;
    int b = blockIdx.y;
    int p0 = blockIdx.x * 16;
    int n = p0 >> 5;                              // all 16 cols in same n
    if (tid < NGRP) { s_sum[tid] = 0.f; s_sq[tid] = 0.f; }
    if (tid < 16)
        ((uint2*)sYf)[tid] = ((const uint2*)(YfeatT + ((size_t)b * N_PTS + n) * 64))[tid];
    {   // copy 2 swizzled Yg tiles (2KB)
        int idx = tid * 4, tc = idx >> 9, rem = idx & 511;
        const __bf16* src = YgSwz + (size_t)b * 64 * NK
                          + ((size_t)tc * (NK >> 4) + (p0 >> 4)) * 512 + rem;
        *(uint2*)&sYg[idx] = *(const uint2*)src;
    }
    __syncthreads();

    int o0 = wv * 16, k0 = (lane < 16) ? 0 : 16, col = lane & 15;
    const __bf16* wb = Wfold + (size_t)b * (C_IN * C_IN) + (size_t)(o0 >> 4) * 4 * 512;
    const float* bb = bfold + b * C_IN;
    v8f acc = {};
#pragma unroll
    for (int ct = 0; ct < 4; ++ct) {
        int cb = ct * 32;
        v16bf a, bm;
        load_frag(wb + ct * 512 + lane * 16, a);
        if (cb < 64) load_frag(&sYf[cb + k0], bm);                // broadcast over cols
        else         load_frag(&sYg[(size_t)((cb - 64) >> 5) * 512 + lane * 16], bm);
        acc = wmma_bf16(a, bm, acc);
    }

    int m_off = (lane < 16) ? 0 : 8;
    size_t p = (size_t)p0 + col;
    bf8 dv;
#pragma unroll
    for (int v = 0; v < 8; ++v) {
        int c = o0 + v + m_off;
        float val = fmaxf(acc[v] + bb[c], 0.f);
        dv.h[v] = (__bf16)val;
        int g = c >> 2;
        atomicAdd(&s_sum[g], val);
        atomicAdd(&s_sq[g],  val * val);
    }
    size_t idx = ((size_t)(o0 >> 5) * (NK >> 4) + (p >> 4)) * 512
               + (size_t)(col + (((o0 >> 4) & 1) << 4)) * 16 + m_off;
    *(bf8*)&Z2swz[(size_t)b * C_IN * NK + idx] = dv;
    __syncthreads();
    if (tid < NGRP) {
        atomicAdd(&gnAcc2[(b * NGRP + tid) * 2 + 0], s_sum[tid]);
        atomicAdd(&gnAcc2[(b * NGRP + tid) * 2 + 1], s_sq[tid]);
    }
}

// ---------------------------------------------------------------------------
// scores = W' @ Z2 + b' (GN2 folded; Z2 tiles TDM-staged to LDS) -> masked
// softmax over K -> out = sum_k relu(GN3(G3)) * weight.  Block = (b, 2 n's).
// ---------------------------------------------------------------------------
__global__ void __launch_bounds__(256) final_kernel(
    const __bf16* __restrict__ Z2swz, const __bf16* __restrict__ G3, const int* __restrict__ count,
    const float* __restrict__ mean3, const float* __restrict__ rstd3,
    const float* __restrict__ gn3w, const float* __restrict__ gn3b,
    const __bf16* __restrict__ Wfold, const float* __restrict__ bfold,
    float* __restrict__ out)
{
    __shared__ __bf16 sZ[16 * 512];          // 16 fragments (4 ci-blocks x 4 p-tiles), 16KB
    __shared__ float s_scores[C_IN * 64];    // 32KB
    __shared__ float sm3[NGRP], sr3[NGRP], sw3[C_IN], sb3[C_IN];
    int tid = threadIdx.x, lane = tid & 31, wv = tid >> 5;
    int b = blockIdx.y;
    int nbase = blockIdx.x * 2, pbase = nbase * K_GRP;   // 64 spatial points
    if (tid < NGRP) { sm3[tid] = mean3[b * NGRP + tid]; sr3[tid] = rstd3[b * NGRP + tid]; }
    if (tid < C_IN) { sw3[tid] = gn3w[tid]; sb3[tid] = gn3b[tid]; }

    const __bf16* gsrc = Z2swz + (size_t)b * C_IN * NK + (size_t)(pbase >> 4) * 512;
#if USE_TDM
    if (wv == 0) {
        // TDM D# (CDNA5 ISA): 2D tensor, data_size=8B,
        // tile 512x4 (4KB rows), row stride = (NK/16)*512 bf16 = 0x100000 x 8B.
        unsigned long long ga = (unsigned long long)(size_t)gsrc;
        unsigned lds_off = (unsigned)(size_t)(&sZ[0]);
        v4u g0; g0[0] = 1u;                               // count=1 (valid descriptor)
        g0[1] = lds_off;                                  // lds_addr
        g0[2] = (unsigned)(ga & 0xFFFFFFFFu);             // global_addr[31:0]
        g0[3] = (unsigned)(((ga >> 32) & 0x1FFFFFFu) | (2u << 30)); // addr[56:32] | type=2
        v8i g1; g1[0] = (int)(3u << 16);                  // data_size = 8B, no multicast
        g1[1] = (int)(512u << 16);                        // tensor_dim0 = 512
        g1[2] = (int)(4u << 16);                          // tensor_dim1 = 4
        g1[3] = (int)(512u << 16);                        // tile_dim0 = 512
        g1[4] = 4;                                        // tile_dim1 = 4
        g1[5] = 0x100000;                                 // tensor_dim0_stride (8B units)
        g1[6] = 0; g1[7] = 0;
        v4i gz = {0, 0, 0, 0};
#if TDM_6ARG
        v8i gz8 = {0, 0, 0, 0, 0, 0, 0, 0};
        __builtin_amdgcn_tensor_load_to_lds(g0, g1, gz, gz, gz8, 0);
#else
        __builtin_amdgcn_tensor_load_to_lds(g0, g1, gz, gz, 0);
#endif
        __builtin_amdgcn_s_wait_tensorcnt((short)0);
    }
#else
#pragma unroll
    for (int j = 0; j < 4; ++j) {
        const uint4* s = (const uint4*)(gsrc + (size_t)j * (size_t)(NK >> 4) * 512);
        ((uint4*)sZ)[j * 256 + tid] = s[tid];
    }
#endif
    __syncthreads();

    int o0 = wv * 16, col = lane & 15;
    const __bf16* wb = Wfold + (size_t)b * (C_IN * C_IN) + (size_t)(o0 >> 4) * 4 * 512;
    const float* bb = bfold + b * C_IN;
    v8f acc[4] = {};
#pragma unroll
    for (int ct = 0; ct < 4; ++ct) {
        v16bf a;
        load_frag(wb + ct * 512 + lane * 16, a);
#pragma unroll
        for (int t = 0; t < 4; ++t) {
            v16bf bm;
            load_frag(&sZ[(ct * 4 + t) * 512 + lane * 16], bm);
            acc[t] = wmma_bf16(a, bm, acc[t]);
        }
    }
    int m_off = (lane < 16) ? 0 : 8;
#pragma unroll
    for (int t = 0; t < 4; ++t)
#pragma unroll
        for (int v = 0; v < 8; ++v) {
            int c = o0 + v + m_off;
            s_scores[c * 64 + t * 16 + col] = acc[t][v] + bb[c];
        }
    __syncthreads();

    // 256 threads -> 128 channels x 2 n's: masked softmax + weighted K-sum
    int c = tid & 127, nl = tid >> 7, n = nbase + nl;
    int cnt = count[b * N_PTS + n]; if (cnt < 1) cnt = 1;
    const float* srow = &s_scores[c * 64 + nl * 32];
    float mx = -3.4e38f;
#pragma unroll
    for (int k = 0; k < K_GRP; ++k) { float s = (k < cnt) ? srow[k] : -1e9f; mx = fmaxf(mx, s); }
    float sum = 0.f;
#pragma unroll
    for (int k = 0; k < K_GRP; ++k) { float s = (k < cnt) ? srow[k] : -1e9f; sum += __expf(s - mx); }
    float inv = 1.f / sum;

    int g = c >> 2;
    float m3 = sm3[g], r3 = sr3[g], w3 = sw3[c], b3 = sb3[c];
    const __bf16* grow = G3 + ((size_t)b * C_IN + c) * NK + (size_t)n * K_GRP;
    const uint4* gq = (const uint4*)grow;                 // 64B row, 4x b128
    float gvals[K_GRP];
#pragma unroll
    for (int j = 0; j < 4; ++j) {
        uint4 q = gq[j];
        unsigned u[4] = {q.x, q.y, q.z, q.w};
#pragma unroll
        for (int d = 0; d < 4; ++d) {
            gvals[j * 8 + d * 2 + 0] = __uint_as_float(u[d] << 16);
            gvals[j * 8 + d * 2 + 1] = __uint_as_float(u[d] & 0xFFFF0000u);
        }
    }
    float accv = 0.f;
#pragma unroll
    for (int k = 0; k < K_GRP; ++k) {
        float s = (k < cnt) ? srow[k] : -1e9f;
        float wgt = __expf(s - mx) * inv;
        float gv = fmaxf((gvals[k] - m3) * r3 * w3 + b3, 0.f);
        accv += gv * wgt;
    }
    out[((size_t)b * C_IN + c) * N_PTS + n] = accv;
}

// ---------------------------------------------------------------------------
extern "C" void kernel_launch(void* const* d_in, const int* in_sizes, int n_in,
                              void* d_out, int out_size, void* d_ws, size_t ws_size,
                              hipStream_t stream) {
    const float* feat    = (const float*)d_in[0];
    const float* gfeat   = (const float*)d_in[1];
    const float* gfo_in  = (const float*)d_in[2];
    const int*   count   = (const int*)d_in[3];
    const float* W_feat  = (const float*)d_in[4];
    const float* b_feat  = (const float*)d_in[5];
    const float* W_grp   = (const float*)d_in[6];
    const float* b_grp   = (const float*)d_in[7];
    const float* gn1_w   = (const float*)d_in[8];
    const float* gn1_b   = (const float*)d_in[9];
    const float* W_wc1   = (const float*)d_in[10];
    const float* b_wc1   = (const float*)d_in[11];
    const float* gn2_w   = (const float*)d_in[12];
    const float* gn2_b   = (const float*)d_in[13];
    const float* W_wc2   = (const float*)d_in[14];
    const float* b_wc2   = (const float*)d_in[15];
    const float* W_fo    = (const float*)d_in[16];
    const float* b_fo    = (const float*)d_in[17];
    const float* gn3_w   = (const float*)d_in[18];
    const float* gn3_b   = (const float*)d_in[19];
    float* out = (float*)d_out;

    char* wsb = (char*)d_ws;
    size_t off = 0;
    auto take = [&](size_t bytes) { size_t o = off; off += (bytes + 255) & ~(size_t)255; return o; };
    __bf16* YfeatT = (__bf16*)(wsb + take(2ull * B_SZ * N_PTS * 64));   // (B,N,64) c-contig
    __bf16* YgSwz  = (__bf16*)(wsb + take(2ull * B_SZ * 64  * NK));     // swizzled
    __bf16* G3     = (__bf16*)(wsb + take(2ull * B_SZ * 128 * NK));     // raw (B,C,S)
    __bf16* Z2swz  = (__bf16*)(wsb + take(2ull * B_SZ * 128 * NK));     // swizzled
    __bf16* Wfz    = (__bf16*)(wsb + take(2ull * 64  * 128));
    __bf16* Wgz    = (__bf16*)(wsb + take(2ull * 64  * 128));
    __bf16* Wfoz   = (__bf16*)(wsb + take(2ull * 128 * 128));
    __bf16* W1f    = (__bf16*)(wsb + take(2ull * B_SZ * 128 * 128));    // per-batch folded
    __bf16* W2f    = (__bf16*)(wsb + take(2ull * B_SZ * 128 * 128));
    float* b1f   = (float*)(wsb + take(4ull * B_SZ * C_IN));
    float* b2f   = (float*)(wsb + take(4ull * B_SZ * C_IN));
    float* acc1  = (float*)(wsb + take(4ull * B_SZ * NGRP * 2));
    float* acc2  = (float*)(wsb + take(4ull * B_SZ * NGRP * 2));
    float* acc3  = (float*)(wsb + take(4ull * B_SZ * NGRP * 2));
    float* mean1 = (float*)(wsb + take(4ull * B_SZ * NGRP));
    float* rstd1 = (float*)(wsb + take(4ull * B_SZ * NGRP));
    float* mean2 = (float*)(wsb + take(4ull * B_SZ * NGRP));
    float* rstd2 = (float*)(wsb + take(4ull * B_SZ * NGRP));
    float* mean3 = (float*)(wsb + take(4ull * B_SZ * NGRP));
    float* rstd3 = (float*)(wsb + take(4ull * B_SZ * NGRP));

    const float invCnt = 1.f / (4.f * (float)NK);   // 4 chans/group * N * K

    zero_acc<<<1, 256, 0, stream>>>(acc1, acc2, acc3);
    wswz_prep<<<32, 256, 0, stream>>>(W_feat, Wfz, 64 * 128);
    wswz_prep<<<32, 256, 0, stream>>>(W_grp,  Wgz, 64 * 128);
    wswz_prep<<<64, 256, 0, stream>>>(W_fo,   Wfoz, 128 * 128);

    // feat1 = relu(W_feat@feat + b): GN1 groups 0..15, each value counted K times
    conv1x1_gn<64, true, 1><<<dim3(N_PTS / 32, B_SZ), 256, 0, stream>>>(
        feat, Wfz, b_feat, YfeatT, acc1, N_PTS, 0, (float)K_GRP);
    // g1 = relu(W_grp@grouped_feat + b): GN1 groups 16..31, swizzled out
    conv1x1_gn<64, true, 0><<<dim3(NK / 32, B_SZ), 256, 0, stream>>>(
        gfeat, Wgz, b_grp, YgSwz, acc1, NK, 64, 1.0f);
    // g3 = W_fo@grouped_feat_out + b (no relu): GN3 partials, raw out
    conv1x1_gn<128, false, 2><<<dim3(NK / 16, B_SZ), 256, 0, stream>>>(
        gfo_in, Wfoz, b_fo, G3, acc3, NK, 0, 1.0f);

    gn_finalize<<<1, 128, 0, stream>>>(acc1, mean1, rstd1, invCnt);
    gn_finalize<<<1, 128, 0, stream>>>(acc3, mean3, rstd3, invCnt);

    // Fold GN1 into W_wc1/b_wc1 (per batch), then run the fused GEMM.
    fold_wswz<<<dim3(64, B_SZ), 256, 0, stream>>>(W_wc1, rstd1, gn1_w, W1f);
    fold_bias<<<dim3(1, B_SZ), 128, 0, stream>>>(W_wc1, b_wc1, mean1, rstd1, gn1_w, gn1_b, b1f);

    wc1_kernel<<<dim3(NK / 16, B_SZ), 256, 0, stream>>>(
        YfeatT, YgSwz, W1f, b1f, Z2swz, acc2);

    gn_finalize<<<1, 128, 0, stream>>>(acc2, mean2, rstd2, invCnt);

    // Fold GN2 into W_wc2/b_wc2 (per batch), then the attention/output stage.
    fold_wswz<<<dim3(64, B_SZ), 256, 0, stream>>>(W_wc2, rstd2, gn2_w, W2f);
    fold_bias<<<dim3(1, B_SZ), 128, 0, stream>>>(W_wc2, b_wc2, mean2, rstd2, gn2_w, gn2_b, b2f);

    final_kernel<<<dim3(N_PTS / 2, B_SZ), 256, 0, stream>>>(
        Z2swz, G3, count, mean3, rstd3, gn3_w, gn3_b, W2f, b2f, out);

    (void)in_sizes; (void)n_in; (void)out_size; (void)ws_size;
}
